// HSN_LSTM_73899207295229
// MI455X (gfx1250) — compile-verified, matrix operand order checked by reading, not run on hardware
//
#include <hip/hip_runtime.h>
#include <hip/hip_bf16.h>

// ---- problem constants (match reference) ----
#define Bsz  64
#define Tlen 128
#define Din  64
#define Nn   128
#define Oo   10
#define HPITCH 136   // bf16 elems per LDS row: 128 + 8 pad (16B-aligned pitch)

typedef __attribute__((ext_vector_type(16))) __bf16 v16bf;
typedef __attribute__((ext_vector_type(8)))  __bf16 v8bf;
typedef __attribute__((ext_vector_type(8)))  float  v8f;

// Load one 16x32 bf16 WMMA operand fragment from an LDS tile whose rows
// (length-128 K vectors, pitch HPITCH) are contiguous in K.
// CDNA5 16-bit operand layout: lanes 0-15 hold K={0..7,16..23} of row `lane`,
// lanes 16-31 hold K={8..15,24..31}.  -> two ds_load_b128 per fragment.
__device__ __forceinline__ v16bf load_frag_bf16(const __bf16* rowp, int kbase, int hi) {
  const __bf16* p0 = rowp + kbase + hi * 8;
  v8bf lo = *reinterpret_cast<const v8bf*>(p0);
  v8bf hh = *reinterpret_cast<const v8bf*>(p0 + 16);
  v16bf r;
#pragma unroll
  for (int e = 0; e < 8; ++e) { r[e] = lo[e]; r[8 + e] = hh[e]; }
  return r;
}

__device__ __forceinline__ float sigmoidf_(float z) { return 1.f / (1.f + __expf(-z)); }

// ---------------------------------------------------------------------------
// Kernel 1: persistent recurrent kernel. grid = D blocks, 512 thr (16 waves).
// Whole T=128 recurrence runs on-chip; weights live in LDS as bf16; gates via
// v_wmma_f32_16x16x32_bf16; membrane/spike/attention streamed in registers.
// Emits hg[b][d][0:128]=attention-avg g_n, hg[b][d][128:256]=h_last.
// ---------------------------------------------------------------------------
__global__ __launch_bounds__(512) void snn_recurrent_kernel(
    const float* __restrict__ x,
    const float* __restrict__ Uj, const float* __restrict__ Ui,
    const float* __restrict__ Uf, const float* __restrict__ Uo,
    const float* __restrict__ Wj, const float* __restrict__ Wi,
    const float* __restrict__ Wf, const float* __restrict__ Wo,
    const float* __restrict__ bj, const float* __restrict__ bi,
    const float* __restrict__ bf_, const float* __restrict__ bo,
    const float* __restrict__ Fa, const float* __restrict__ Fab,
    float* __restrict__ hg)
{
  extern __shared__ unsigned char smem[];
  __bf16* Wt   = reinterpret_cast<__bf16*>(smem);     // [4][128][HPITCH] transposed W (B operand rows = output col)
  __bf16* hbf  = Wt + 4 * Nn * HPITCH;                // [64][HPITCH] current h (A operand)
  float*  fb   = reinterpret_cast<float*>(hbf + Bsz * HPITCH);
  float* xlds   = fb;               // [64]   x[:, t, d]
  float* Ul     = xlds + Bsz;       // [4][128]
  float* Bl     = Ul + 4 * Nn;      // [4][128]
  float* Fal    = Bl + 4 * Nn;      // [128]
  float* alphal = Fal + Nn;         // [64]
  float* denl   = alphal + Bsz;     // [64]
  float* fabl   = denl + Bsz;       // [1]

  const int d   = blockIdx.x;
  const int tid = threadIdx.x;

  // ---- one-time staging: transpose W[d] (k,m)->(m,k) into bf16 LDS ----
  const float* Wg[4] = { Wj + (size_t)d * Nn * Nn, Wi + (size_t)d * Nn * Nn,
                         Wf + (size_t)d * Nn * Nn, Wo + (size_t)d * Nn * Nn };
#pragma unroll
  for (int g = 0; g < 4; ++g)
    for (int idx = tid; idx < Nn * Nn; idx += 512) {
      const int k = idx >> 7, m = idx & (Nn - 1);            // coalesced global read
      Wt[(g * Nn + m) * HPITCH + k] = (__bf16)Wg[g][idx];
    }
  if (tid < Nn) {
    Ul[0 * Nn + tid] = Uj[(size_t)d * Nn + tid];
    Ul[1 * Nn + tid] = Ui[(size_t)d * Nn + tid];
    Ul[2 * Nn + tid] = Uf[(size_t)d * Nn + tid];
    Ul[3 * Nn + tid] = Uo[(size_t)d * Nn + tid];
    Bl[0 * Nn + tid] = bj[(size_t)d * Nn + tid];
    Bl[1 * Nn + tid] = bi[(size_t)d * Nn + tid];
    Bl[2 * Nn + tid] = bf_[(size_t)d * Nn + tid];
    Bl[3 * Nn + tid] = bo[(size_t)d * Nn + tid];
    Fal[tid] = Fa[(size_t)d * Nn + tid];
  }
  if (tid == 0) fabl[0] = Fab[d];
  for (int idx = tid; idx < Bsz * HPITCH; idx += 512) hbf[idx] = (__bf16)0.f;
  __syncthreads();

  // ---- per-lane tile ownership ----
  const int wave = tid >> 5, lane = tid & 31;
  const int mt = wave & 3;          // M-tile (batch rows 16*mt..)
  const int ng = wave >> 2;         // N-tile-pair (cols 32*ng..)
  const int nl = lane & 15, hi = lane >> 4;
  const int bbase  = mt * 16;
  const int nbase0 = ng * 32;

  float h_reg[2][8], c_reg[2][8], accn[2][8];
#pragma unroll
  for (int jj = 0; jj < 2; ++jj)
#pragma unroll
    for (int e = 0; e < 8; ++e) { h_reg[jj][e] = 0.f; c_reg[jj][e] = 0.f; accn[jj][e] = 0.f; }

  float Ur[4][2], Brr[4][2];
#pragma unroll
  for (int g = 0; g < 4; ++g) {
    Ur[g][0]  = Ul[g * Nn + nbase0 + nl];       Ur[g][1]  = Ul[g * Nn + nbase0 + 16 + nl];
    Brr[g][0] = Bl[g * Nn + nbase0 + nl];       Brr[g][1] = Bl[g * Nn + nbase0 + 16 + nl];
  }
  float den_acc = 0.f;
  const float DT = 0.01f;
  const float A_OUT = 0.99004983374916811f;     // exp(-0.01)

  for (int t = 0; t < Tlen; ++t) {
    if (tid < Bsz) xlds[tid] = x[((size_t)tid * Tlen + t) * Din + d];

    // ---- 4-gate GEMM: h[64x128] @ W_g[128x128], bf16 WMMA, f32 acc ----
    v8f acc[4][2];
#pragma unroll
    for (int g = 0; g < 4; ++g)
#pragma unroll
      for (int jj = 0; jj < 2; ++jj)
#pragma unroll
        for (int e = 0; e < 8; ++e) acc[g][jj][e] = 0.f;

#pragma unroll
    for (int kk = 0; kk < 4; ++kk) {
      const v16bf a = load_frag_bf16(hbf + (bbase + nl) * HPITCH, kk * 32, hi);  // A shared by gates
#pragma unroll
      for (int jj = 0; jj < 2; ++jj) {
        const int ncol = nbase0 + jj * 16 + nl;
#pragma unroll
        for (int g = 0; g < 4; ++g) {
          const v16bf bmat = load_frag_bf16(Wt + ((size_t)g * Nn + ncol) * HPITCH, kk * 32, hi);
          acc[g][jj] = __builtin_amdgcn_wmma_f32_16x16x32_bf16(
              false, a, false, bmat, (short)0, acc[g][jj], false, false);
        }
      }
    }
    __syncthreads();  // all waves done reading hbf; xlds staged

    // ---- elementwise membrane / spike update on owned C elements ----
#pragma unroll
    for (int jj = 0; jj < 2; ++jj) {
      const int n = nbase0 + jj * 16 + nl;
#pragma unroll
      for (int e = 0; e < 8; ++e) {
        const int b = bbase + hi * 8 + e;
        const float xv = xlds[b];
        const float jg = tanhf   (acc[0][jj][e] + xv * Ur[0][jj] + Brr[0][jj]);
        const float ig = sigmoidf_(acc[1][jj][e] + xv * Ur[1][jj] + Brr[1][jj]);
        const float fg = sigmoidf_(acc[2][jj][e] + xv * Ur[2][jj] + Brr[2][jj]);
        const float og = sigmoidf_(acc[3][jj][e] + xv * Ur[3][jj] + Brr[3][jj]);
        const float alpha = __expf(-0.00125f   * jg);     // -(dt/tau_m)*0.5
        const float ro    = __expf(-1.5625e-4f * ig);     // -(dt/tau_adp)*0.5
        const float badp  = ro * 0.1f + (1.f - ro) * ig;
        const float Bth   = 0.04f + 1.8f * badp;
        const float mem   = jg * alpha + (1.f - alpha) * h_reg[jj][e] - Bth * ig * DT;
        const float spk   = (mem - Bth) > 0.f ? 1.f : 0.f;  // heaviside forward
        const float mem_o = mem * A_OUT + (1.f - A_OUT) * spk + 0.08f;
        const float cnew  = c_reg[jj][e] * fg + ig * spk * mem_o;
        const float hnew  = og * tanhf(cnew);
        c_reg[jj][e] = cnew;
        h_reg[jj][e] = hnew;
        hbf[b * HPITCH + n] = (__bf16)hnew;               // publish for next GEMM + alpha dot
      }
    }
    __syncthreads();  // h_t+1 visible

    // ---- streamed temporal attention: alpha[b] = exp(tanh(h . F_a + b_a)) ----
    if (tid < Bsz) {
      float s = fabl[0];
      const __bf16* hp = hbf + tid * HPITCH;
#pragma unroll 8
      for (int n2 = 0; n2 < Nn; ++n2) s += (float)hp[n2] * Fal[n2];
      const float av = __expf(tanhf(s));
      alphal[tid] = av;
      den_acc += av;
    }
    __syncthreads();  // alphas broadcast

#pragma unroll
    for (int jj = 0; jj < 2; ++jj)
#pragma unroll
      for (int e = 0; e < 8; ++e)
        accn[jj][e] += alphal[bbase + hi * 8 + e] * h_reg[jj][e];
  }

  if (tid < Bsz) denl[tid] = den_acc;
  __syncthreads();

  // ---- emit hg[b][d][0:128]=g_n, [128:256]=h_last ----
#pragma unroll
  for (int jj = 0; jj < 2; ++jj) {
    const int n = nbase0 + jj * 16 + nl;
#pragma unroll
    for (int e = 0; e < 8; ++e) {
      const int b = bbase + hi * 8 + e;
      float* outp = hg + ((size_t)b * Din + d) * (2 * Nn);
      outp[n]      = accn[jj][e] / denl[b];
      outp[Nn + n] = h_reg[jj][e];
    }
  }
}

// ---------------------------------------------------------------------------
// Kernel 2: tiny head. grid = B blocks, 64 thr (thread = d).
// mu = hg @ Phi_w^T + Phi_b ; beta = softmax-over-D(exp(tanh(hg.Fb+b))),
// beta = 0.01*(-log2 beta) ; out[b,o] = sum_d beta_d * mu_{d,o}.
// ---------------------------------------------------------------------------
__global__ __launch_bounds__(64) void snn_head_kernel(
    const float* __restrict__ hg, const float* __restrict__ Fbw,
    const float* __restrict__ Fbb, const float* __restrict__ Phiw,
    const float* __restrict__ Phib, float* __restrict__ out)
{
  __shared__ float brs[Din];
  __shared__ float mus[Din][12];
  __shared__ float betal[Din];
  __shared__ float ssum;
  const int b  = blockIdx.x;
  const int dd = threadIdx.x;
  const float* row = hg + ((size_t)b * Din + dd) * (2 * Nn);
  float zb = Fbb[0];
  float mu[Oo];
#pragma unroll
  for (int o = 0; o < Oo; ++o) mu[o] = Phib[o];
  for (int k = 0; k < 2 * Nn; ++k) {
    const float v = row[k];
    zb += v * Fbw[k];
#pragma unroll
    for (int o = 0; o < Oo; ++o) mu[o] += v * Phiw[o * (2 * Nn) + k];
  }
  brs[dd] = __expf(tanhf(zb));
#pragma unroll
  for (int o = 0; o < Oo; ++o) mus[dd][o] = mu[o];
  __syncthreads();
  if (dd == 0) {
    float s = 0.f;
    for (int q = 0; q < Din; ++q) s += brs[q];
    ssum = s;
  }
  __syncthreads();
  betal[dd] = 0.01f * (__log2f(ssum) - __log2f(brs[dd]));  // 0.01*(-log2(br/sum))
  __syncthreads();
  if (dd < Oo) {
    float acc = 0.f;
    for (int q = 0; q < Din; ++q) acc += betal[q] * mus[q][dd];
    out[(size_t)b * Oo + dd] = acc;
  }
}

extern "C" void kernel_launch(void* const* d_in, const int* in_sizes, int n_in,
                              void* d_out, int out_size, void* d_ws, size_t ws_size,
                              hipStream_t stream) {
  (void)in_sizes; (void)n_in; (void)out_size; (void)ws_size;
  const float* x   = (const float*)d_in[0];
  const float* Uj  = (const float*)d_in[1];
  const float* Ui  = (const float*)d_in[2];
  const float* Uf  = (const float*)d_in[3];
  const float* Uo  = (const float*)d_in[4];
  const float* Wj  = (const float*)d_in[5];
  const float* Wi  = (const float*)d_in[6];
  const float* Wf  = (const float*)d_in[7];
  const float* Wo  = (const float*)d_in[8];
  const float* bj  = (const float*)d_in[9];
  const float* bi  = (const float*)d_in[10];
  const float* bf_ = (const float*)d_in[11];
  const float* bo  = (const float*)d_in[12];
  const float* Fa  = (const float*)d_in[13];
  const float* Fab = (const float*)d_in[14];
  const float* Fbw = (const float*)d_in[15];
  const float* Fbb = (const float*)d_in[16];
  const float* Phw = (const float*)d_in[17];
  const float* Phb = (const float*)d_in[18];

  float* hg = (float*)d_ws;  // [B][D][2N] f32 = 4 MB scratch

  const size_t shmem = (size_t)(4 * Nn * HPITCH + Bsz * HPITCH) * sizeof(__bf16)
                     + (size_t)(Bsz + 4 * Nn + 4 * Nn + Nn + Bsz + Bsz + 4) * sizeof(float);
  // opt in to >64KB dynamic LDS (320 KB/WGP on CDNA5); ignore error if already set
  (void)hipFuncSetAttribute((const void*)snn_recurrent_kernel,
                            hipFuncAttributeMaxDynamicSharedMemorySize, (int)shmem);

  snn_recurrent_kernel<<<Din, 512, shmem, stream>>>(
      x, Uj, Ui, Uf, Uo, Wj, Wi, Wf, Wo, bj, bi, bf_, bo, Fa, Fab, hg);
  snn_head_kernel<<<Bsz, 64, 0, stream>>>(hg, Fbw, Fbb, Phw, Phb, (float*)d_out);
}
